// AxialAttention3D_23390391894618
// MI455X (gfx1250) — matrix operand order repaired and checked by reading, not compiled, not run on hardware
//
#include <hip/hip_runtime.h>
#include <hip/hip_bf16.h>

// ---------------------------------------------------------------------------
// AxialAttention3D for MI455X (gfx1250).
// B=1, L=13, H=48, W=96, D=512, heads=16, hd=32, Ntok = 59904.
// GEMMs in bf16 WMMA (v_wmma_f32_16x16x32_bf16) with f32 accumulation.
// Fragment operands staged in LDS so every fragment load is ds_load_b128.
// Attention (S<=96) exact in fp32 VALU with online softmax + exact rotary.
// ---------------------------------------------------------------------------

typedef __attribute__((ext_vector_type(16))) __bf16          v16bf;
typedef __attribute__((ext_vector_type(8)))  unsigned short  us8;
typedef __attribute__((ext_vector_type(8)))  float           v8f;

struct Frag16 { us8 lo; us8 hi; };   // 32 bytes -> bit_cast to v16bf

#define NTOK  59904
#define DMODEL 512
#define PI_F 3.14159265358979323846f
#define TWO_PI_F 6.28318530717958647692f

__device__ __forceinline__ unsigned short f2bf(float f) {
    unsigned int u = __builtin_bit_cast(unsigned int, f);
    unsigned int r = u + 0x7FFFu + ((u >> 16) & 1u);        // round-to-nearest-even
    if ((u & 0x7F800000u) == 0x7F800000u) r = u;            // inf / NaN: truncate
    return (unsigned short)(r >> 16);
}

// ---------------------------------------------------------------------------
// f32 -> bf16 elementwise (weights)
// ---------------------------------------------------------------------------
__global__ void cvt_f32_bf16_kernel(const float* __restrict__ src,
                                    unsigned short* __restrict__ dst, int n) {
    int i = blockIdx.x * blockDim.x + threadIdx.x;
    if (i < n) dst[i] = f2bf(src[i]);
}

// ---------------------------------------------------------------------------
// LayerNorm over D=512 per token, output bf16. One wave32 per row.
// blockDim = 256 (8 waves); grid = NTOK/8.
// ---------------------------------------------------------------------------
__global__ __launch_bounds__(256) void ln_bf16_kernel(
    const float* __restrict__ x, const float* __restrict__ g,
    const float* __restrict__ b, unsigned short* __restrict__ out) {
    int wave = threadIdx.x >> 5;
    int lane = threadIdx.x & 31;
    int row  = blockIdx.x * 8 + wave;
    const float* xr = x + (size_t)row * DMODEL;
    float v[16];
    float s = 0.f, s2 = 0.f;
#pragma unroll
    for (int i = 0; i < 16; i++) {
        float t = xr[lane + 32 * i];
        v[i] = t; s += t; s2 += t * t;
    }
#pragma unroll
    for (int off = 16; off >= 1; off >>= 1) {
        s  += __shfl_xor(s,  off, 32);
        s2 += __shfl_xor(s2, off, 32);
    }
    float mean = s * (1.f / 512.f);
    float var  = s2 * (1.f / 512.f) - mean * mean;
    float rs   = rsqrtf(var + 1e-5f);
    unsigned short* orow = out + (size_t)row * DMODEL;
#pragma unroll
    for (int i = 0; i < 16; i++) {
        int d = lane + 32 * i;
        orow[d] = f2bf((v[i] - mean) * rs * g[d] + b[d]);
    }
}

// ---------------------------------------------------------------------------
// bf16 GEMM:  D[M,N] = A[M,K](bf16) x B[K,N](bf16) + bias[N] (+ resid[M,N])
// Tiles: 128x128x32, 256 threads = 8 waves, wave computes 32x64 (2x4 frags).
// A staged row-major (stride 40), B staged TRANSPOSED (BsT[n][k], stride 40)
// so every fragment load is two 16B contiguous LDS reads (ds_load_b128).
// Requires M%128==0, N%128==0, K%32==0 (all satisfied here).
// ---------------------------------------------------------------------------
template <bool HAS_RES>
__global__ __launch_bounds__(256) void gemm_bf16_kernel(
    const unsigned short* __restrict__ A,   // [M,K] row-major bf16
    const unsigned short* __restrict__ B,   // [K,N] row-major bf16
    const float* __restrict__ bias,         // [N]
    const float* __restrict__ resid,        // [M,N] (used iff HAS_RES)
    float* __restrict__ D, int M, int N, int K) {
    const int LDT = 40;    // padded stride (ushort units), 80B rows
    __shared__ unsigned short As [128 * 40];   // As [m][k]
    __shared__ unsigned short BsT[128 * 40];   // BsT[n][k]

    int tid  = threadIdx.x;
    int lane = tid & 31;
    int wid  = tid >> 5;
    int wm   = wid & 3;     // 32-row strip within 128
    int wn   = wid >> 2;    // 64-col strip within 128
    int rowBase = blockIdx.y * 128;
    int colBase = blockIdx.x * 128;

    v8f acc[2][4];
#pragma unroll
    for (int i = 0; i < 2; i++)
#pragma unroll
        for (int j = 0; j < 4; j++)
#pragma unroll
            for (int e = 0; e < 8; e++) acc[i][j][e] = 0.f;

    int lm   = lane & 15;
    int kb   = (lane >> 4) * 8;    // A fragment k-base (16-bit A 16x32 layout)
    int kb16 = (lane >> 4) * 16;   // unused; B fragments also load at kb

    // loop-invariant per-thread prefetch bases (32-bit offsets added per iter)
    const unsigned short* pfA = A + (size_t)(rowBase + (tid >> 4)) * K + (tid & 15) * 2;
    const unsigned short* pfB = B + (size_t)(tid >> 6) * N + colBase + (tid & 63) * 2;

    for (int k0 = 0; k0 < K; k0 += 32) {
        // branchless prefetch of next k-tile (global_prefetch_b8), clamped in-bounds
        int kp = min(k0 + 32, K - 32);
        __builtin_prefetch(pfA + kp, 0, 1);
        __builtin_prefetch(pfB + kp * N, 0, 1);

        // ---- stage A tile (128x32) coalesced as dwords, row-major ----
#pragma unroll
        for (int r = 0; r < 8; r++) {
            int i   = tid + 256 * r;          // 0..2047 dwords
            int row = i >> 4;
            int c2  = (i & 15) * 2;
            unsigned int val = *(const unsigned int*)(A + (size_t)(rowBase + row) * K + k0 + c2);
            *(unsigned int*)(As + row * LDT + c2) = val;
        }
        // ---- stage B tile (32x128) TRANSPOSED: 2x2 element blocks ----
#pragma unroll
        for (int r = 0; r < 4; r++) {
            int u  = tid + 256 * r;           // 0..1023 blocks
            int k  = (u >> 6) * 2;            // 0..30
            int n2 = (u & 63) * 2;            // 0..126
            unsigned int a0 = *(const unsigned int*)(B + (size_t)(k0 + k)     * N + colBase + n2);
            unsigned int a1 = *(const unsigned int*)(B + (size_t)(k0 + k + 1) * N + colBase + n2);
            // (k,n2),(k,n2+1) in a0 ; (k+1,n2),(k+1,n2+1) in a1
            unsigned int c0 = (a0 & 0xFFFFu) | (a1 << 16);          // column n2:   k, k+1
            unsigned int c1 = (a0 >> 16)     | (a1 & 0xFFFF0000u);  // column n2+1: k, k+1
            *(unsigned int*)(BsT +  n2      * LDT + k) = c0;
            *(unsigned int*)(BsT + (n2 + 1) * LDT + k) = c1;
        }
        __syncthreads();

        // ---- A fragments: lane(0-15)=rows m, element e -> K = kb + (e<8?e:e+8)
        //      => two contiguous 16B runs at [m][kb] and [m][kb+16]
        v16bf af[2];
#pragma unroll
        for (int i = 0; i < 2; i++) {
            int m = wm * 32 + i * 16 + lm;
            Frag16 f;
            f.lo = *(const us8*)(As + m * LDT + kb);
            f.hi = *(const us8*)(As + m * LDT + kb + 16);
            af[i] = __builtin_bit_cast(v16bf, f);
        }
        // ---- B fragments: element e -> K = kb + (e<8?e:e+8), matching A
        v16bf bfr[4];
#pragma unroll
        for (int j = 0; j < 4; j++) {
            int nn = wn * 64 + j * 16 + lm;
            Frag16 f;
            f.lo = *(const us8*)(BsT + nn * LDT + kb);
            f.hi = *(const us8*)(BsT + nn * LDT + kb + 16);
            bfr[j] = __builtin_bit_cast(v16bf, f);
        }
#pragma unroll
        for (int i = 0; i < 2; i++)
#pragma unroll
            for (int j = 0; j < 4; j++)
                acc[i][j] = __builtin_amdgcn_wmma_f32_16x16x32_bf16(
                    false, af[i], false, bfr[j], (short)0, acc[i][j], false, false);
        __syncthreads();
    }

    // ---- epilogue: C/D layout VGPR r -> M = (lane>>4)*8 + r, N = lane&15
    int mb = (lane >> 4) * 8;
#pragma unroll
    for (int j = 0; j < 4; j++) {
        int col = colBase + wn * 64 + j * 16 + lm;
        float bv = bias[col];
#pragma unroll
        for (int i = 0; i < 2; i++) {
            int gm0 = rowBase + wm * 32 + i * 16 + mb;
#pragma unroll
            for (int r = 0; r < 8; r++) {
                int idx = (gm0 + r) * N + col;       // < 2^31 for all our shapes
                float v = acc[i][j][r] + bv;
                if (HAS_RES) v += resid[idx];
                D[idx] = v;
            }
        }
    }
}

// ---------------------------------------------------------------------------
// Attention per (n, head). qkv: [Ntok, 1536] f32 in natural token order,
// token t = (n/nlc)*sa + (n%nlc)*sb + s*ss. Exact rotary (mode 0=lat,
// 1=lon, 2=none), online-softmax, output bf16 into obf[t*512 + h*32 + d].
// blockDim = 128, grid = (Nbatch, 16).
// ---------------------------------------------------------------------------
#define MAX_S 96
__global__ __launch_bounds__(128) void attn_kernel(
    const float* __restrict__ qkv, unsigned short* __restrict__ obf,
    const float* __restrict__ grid, int mode,
    int S, int nlc, int sa, int sb, int ss) {
    __shared__ float sq[MAX_S * 32];
    __shared__ float sk[MAX_S * 32];
    __shared__ float sv[MAX_S * 32];

    int n = blockIdx.x, h = blockIdx.y, tid = threadIdx.x;
    long long tb = (long long)(n / nlc) * sa + (long long)(n % nlc) * sb;

    for (int idx = tid; idx < S * 32; idx += blockDim.x) {
        int s = idx >> 5, d = idx & 31;
        size_t base = (size_t)(tb + (long long)s * ss) * 1536 + h * 32 + d;
        sq[idx] = qkv[base];
        sk[idx] = qkv[base + 512];
        sv[idx] = qkv[base + 1024];
    }
    __syncthreads();

    if (mode < 2) {
        int s = tid;
        if (s < S) {
            float p;
            if (mode == 0) {   // lat: phase = clip(lat_grid[s])
                const float lim = 1.5707963267948966f - 1e-6f;
                p = fminf(fmaxf(grid[s], -lim), lim);
            } else {           // lon: phase = remainder(lon+pi, 2pi) - pi
                float v = grid[s] + PI_F;
                float r = fmodf(v, TWO_PI_F);
                if (r < 0.f) r += TWO_PI_F;
                p = r - PI_F;
            }
            float sn[16], cs[16];
#pragma unroll
            for (int j = 0; j < 16; j++) {
                // inv_freq = 10000^(-j/16)
                float invf = expf(-(float)j * 0.5756462732485115f);
                sincosf(p * invf, &sn[j], &cs[j]);
            }
            float xr[32];
#pragma unroll
            for (int d = 0; d < 32; d++) xr[d] = sq[s * 32 + d];
#pragma unroll
            for (int j = 0; j < 16; j++) {
                sq[s * 32 + j]      = xr[2 * j] * cs[j] - xr[2 * j + 1] * sn[j];
                sq[s * 32 + 16 + j] = xr[2 * j] * sn[j] + xr[2 * j + 1] * cs[j];
            }
#pragma unroll
            for (int d = 0; d < 32; d++) xr[d] = sk[s * 32 + d];
#pragma unroll
            for (int j = 0; j < 16; j++) {
                sk[s * 32 + j]      = xr[2 * j] * cs[j] - xr[2 * j + 1] * sn[j];
                sk[s * 32 + 16 + j] = xr[2 * j] * sn[j] + xr[2 * j + 1] * cs[j];
            }
        }
        __syncthreads();
    }

    int s = tid;
    if (s < S) {
        float q[32];
#pragma unroll
        for (int d = 0; d < 32; d++) q[d] = sq[s * 32 + d];
        float m = -3.4e38f, l = 0.f, acc[32];
#pragma unroll
        for (int d = 0; d < 32; d++) acc[d] = 0.f;
        const float scale = 0.17677669529663689f;  // 1/sqrt(32)
        for (int j = 0; j < S; j++) {
            float dot = 0.f;
#pragma unroll
            for (int d = 0; d < 32; d++) dot += q[d] * sk[j * 32 + d];
            float sc = dot * scale;
            float nm = fmaxf(m, sc);
            float f  = expf(m - nm);
            float e  = expf(sc - nm);
            l = l * f + e;
#pragma unroll
            for (int d = 0; d < 32; d++) acc[d] = acc[d] * f + e * sv[j * 32 + d];
            m = nm;
        }
        float rl = 1.f / l;
        unsigned short* orow =
            obf + (size_t)(tb + (long long)s * ss) * DMODEL + h * 32;
#pragma unroll
        for (int d = 0; d < 32; d++) orow[d] = f2bf(acc[d] * rl);
    }
}

// ---------------------------------------------------------------------------
// Host-side orchestration
// ---------------------------------------------------------------------------
extern "C" void kernel_launch(void* const* d_in, const int* in_sizes, int n_in,
                              void* d_out, int out_size, void* d_ws, size_t ws_size,
                              hipStream_t stream) {
    const int L = 13, H = 48, W = 96;
    const int Ntok = NTOK;

    const float* x         = (const float*)d_in[0];
    const float* lat_grid  = (const float*)d_in[1];
    const float* lon_grid  = (const float*)d_in[2];
    const float* lat_qkv_w = (const float*)d_in[3];
    const float* lat_qkv_b = (const float*)d_in[4];
    const float* lon_qkv_w = (const float*)d_in[5];
    const float* lon_qkv_b = (const float*)d_in[6];
    const float* lev_qkv_w = (const float*)d_in[7];
    const float* lev_qkv_b = (const float*)d_in[8];
    const float* proj_w    = (const float*)d_in[9];
    const float* proj_b    = (const float*)d_in[10];
    const float* g_lat = (const float*)d_in[11];
    const float* b_lat = (const float*)d_in[12];
    const float* g_lon = (const float*)d_in[13];
    const float* b_lon = (const float*)d_in[14];
    const float* g_lev = (const float*)d_in[15];
    const float* b_lev = (const float*)d_in[16];

    // Workspace layout
    char* ws = (char*)d_ws;
    size_t off = 0;
    float* cur = (float*)(ws + off);                   off += (size_t)Ntok * 512 * 4;  // 122.7 MB
    unsigned short* xn  = (unsigned short*)(ws + off); off += (size_t)Ntok * 512 * 2;  // 61.3 MB
    float* qkv = (float*)(ws + off);                   off += (size_t)Ntok * 1536 * 4; // 368 MB
    unsigned short* obf = (unsigned short*)(ws + off); off += (size_t)Ntok * 512 * 2;  // 61.3 MB
    unsigned short* wbf  = (unsigned short*)(ws + off); off += (size_t)512 * 1536 * 2;
    unsigned short* pwbf = (unsigned short*)(ws + off); off += (size_t)512 * 512 * 2;

    hipMemcpyAsync(cur, x, (size_t)Ntok * 512 * 4, hipMemcpyDeviceToDevice, stream);

    // Convert projection weight once
    {
        int nw = 512 * 512;
        cvt_f32_bf16_kernel<<<(nw + 255) / 256, 256, 0, stream>>>(proj_w, pwbf, nw);
    }

    struct Blk {
        const float *qw, *qb, *g, *b, *grid;
        int mode, S, Nb, nlc, sa, sb, ss;
    };
    const Blk blks[3] = {
        // lat: seq over H; n = l*W + w; T = l*H*W + s*W + w
        { lat_qkv_w, lat_qkv_b, g_lat, b_lat, lat_grid, 0, H, L * W, W, H * W, 1, W },
        // lon: seq over W; n = l*H + h; T = n*W + s
        { lon_qkv_w, lon_qkv_b, g_lon, b_lon, lon_grid, 1, W, L * H, 1, W, 0, 1 },
        // lev: seq over L; n = h*W + w; T = n + s*H*W
        { lev_qkv_w, lev_qkv_b, g_lev, b_lev, nullptr, 2, L, H * W, 1, 1, 0, H * W },
    };

    for (int bi = 0; bi < 3; bi++) {
        const Blk& bk = blks[bi];
        // qkv weight -> bf16
        int nw = 512 * 1536;
        cvt_f32_bf16_kernel<<<(nw + 255) / 256, 256, 0, stream>>>(bk.qw, wbf, nw);
        // LayerNorm -> bf16 (natural token order)
        ln_bf16_kernel<<<Ntok / 8, 256, 0, stream>>>(cur, bk.g, bk.b, xn);
        // QKV GEMM: [Ntok,512] x [512,1536]
        gemm_bf16_kernel<false><<<dim3(1536 / 128, Ntok / 128), 256, 0, stream>>>(
            xn, wbf, bk.qb, nullptr, qkv, Ntok, 1536, 512);
        // Attention per (n, head)
        attn_kernel<<<dim3(bk.Nb, 16), 128, 0, stream>>>(
            qkv, obf, bk.grid, bk.mode, bk.S, bk.nlc, bk.sa, bk.sb, bk.ss);
        // Projection GEMM + bias + residual; last block writes d_out
        float* dst = (bi == 2) ? (float*)d_out : cur;
        gemm_bf16_kernel<true><<<dim3(512 / 128, Ntok / 128), 256, 0, stream>>>(
            obf, pwbf, proj_b, cur, dst, Ntok, 512, 512);
    }
}
